// _RFCN_68101001445378
// MI455X (gfx1250) — matrix operand order repaired and checked by reading, compiled, not run on hardware
//
#include <hip/hip_runtime.h>
#include <math.h>

// ---------------- problem constants (from reference) ----------------
#define POOLP   7
#define NCLS    21
#define NBOX    4
#define NOUT    25          // 21 cls + 4 bbox outputs, padded to 32 in WMMA
#define B_      4
#define C_      512
#define H_      96
#define W_      96
#define R_      300
#define NROI    (B_ * R_)   // 1200
#define HP      97          // padded integral dim
#define PLANES  (B_ * C_)   // 2048
#define PSTRIDE 2048        // S_t innermost stride (b*512+c)
#define KTOT    (49 * 512)  // 25088 flattened (bin, channel) reduction dim
#define NCHUNK  (KTOT / 32) // 784 K-chunks of 32

typedef __attribute__((ext_vector_type(16))) _Float16 v16h;
typedef __attribute__((ext_vector_type(8)))  float    v8f;

// ---------------- helpers: aligned vector loads ----------------
__device__ __forceinline__ void ld8(const float* __restrict__ p, float* o) {
  float4 x = *(const float4*)p;
  float4 y = *(const float4*)(p + 4);
  o[0]=x.x; o[1]=x.y; o[2]=x.z; o[3]=x.w;
  o[4]=y.x; o[5]=y.y; o[6]=y.z; o[7]=y.w;
}
__device__ __forceinline__ void ld16(const float* __restrict__ p, float* o) {
  ld8(p, o); ld8(p + 8, o + 8);
}

// ============================================================
// Kernel 1: per-(b,c) 96x96 integral image -> transposed S_t
// S_t[(h*97+w)*2048 + b*512 + c], rows/cols padded with zeros.
// One block per plane; plane lives in LDS for the 2-pass scan.
// Scattered 4B stores merge in the 192MB L2 before writeback.
// ============================================================
__global__ __launch_bounds__(128)
void rfcn_integral_kernel(const float* __restrict__ feat, float* __restrict__ St) {
  __shared__ float tile[H_][W_ + 1];     // +1 pad vs bank conflicts on row scan
  const int plane = blockIdx.x;          // b*512 + c
  const float* src = feat + (size_t)plane * (H_ * W_);

  for (int i = threadIdx.x; i < H_ * W_; i += 128)
    tile[i / W_][i % W_] = src[i];
  __syncthreads();

  for (int r = threadIdx.x; r < H_; r += 128) {        // row prefix (axis=3)
    float s = 0.f;
    for (int w = 0; w < W_; ++w) { s += tile[r][w]; tile[r][w] = s; }
  }
  __syncthreads();

  for (int c = threadIdx.x; c < W_; c += 128) {        // col prefix (axis=2)
    float s = 0.f;
    for (int h = 0; h < H_; ++h) { s += tile[h][c]; tile[h][c] = s; }
  }
  __syncthreads();

  for (int i = threadIdx.x; i < HP * HP; i += 128) {
    int h = i / HP, w = i % HP;
    float v = (h == 0 || w == 0) ? 0.f : tile[h - 1][w - 1];
    St[(size_t)i * PSTRIDE + plane] = v;
  }
}

// ============================================================
// Kernel 2: fused PSROI gather + WMMA GEMM + bias + softmax.
// Block = 16 rois x 25 outputs; 8 waves split K=25088; each wave
// runs two v_wmma_f32_16x16x32_f16 per 32-wide K chunk.
// Corner offsets packed as int4 in LDS (1x ds_load_b128/chunk);
// weight addresses linear in k (bases hoisted); pad columns 25..31
// load a clamped duplicate row (never read in epilogue) -> no
// per-iteration masking VALU.
// ============================================================
__global__ __launch_bounds__(256)
void rfcn_psroi_wmma_kernel(const float* __restrict__ St,
                            const float* __restrict__ rois,
                            const float* __restrict__ Wc,
                            const float* __restrict__ bc,
                            const float* __restrict__ Wb,
                            const float* __restrict__ bb,
                            float* __restrict__ out) {
  // per-(roi,bin) packed corner element offsets into S_t (incl. batch base)
  __shared__ int4  g_off[16][49];         // {o00, o01, o10, o11}
  __shared__ float g_inv[16][49];
  __shared__ float red[8 * 2 * 8 * 32];   // 8 waves x 2 halves x v8f x 32 lanes
  __shared__ float score[16][32];

  const int tid  = threadIdx.x;
  const int roi0 = blockIdx.x * 16;

  // ---- per-roi, per-bin geometry (R-FCN PSROI semantics) ----
  for (int t = tid; t < 16 * 49; t += 256) {
    int r = t / 49, bin = t % 49;
    const float* rr = rois + (size_t)(roi0 + r) * 5;
    const int bbase = (int)rr[0] * C_;
    float x1 = rintf(rr[1]) * 0.0625f;
    float y1 = rintf(rr[2]) * 0.0625f;
    float x2 = rintf(rr[3] + 1.f) * 0.0625f;
    float y2 = rintf(rr[4] + 1.f) * 0.0625f;
    float bw = fmaxf(x2 - x1, 0.1f) * (1.f / POOLP);
    float bh = fmaxf(y2 - y1, 0.1f) * (1.f / POOLP);
    int ph = bin / 7, pw = bin % 7;
    int hs = min(max((int)floorf((float)ph * bh + y1), 0), H_);
    int he = min(max((int)ceilf(((float)ph + 1.f) * bh + y1), 0), H_);
    int ws = min(max((int)floorf((float)pw * bw + x1), 0), W_);
    int we = min(max((int)ceilf(((float)pw + 1.f) * bw + x1), 0), W_);
    // element offsets: (h*HP + w)*PSTRIDE + b*C  (max ~19.3M, fits i32)
    int4 o;
    o.x = (hs * HP + ws) * PSTRIDE + bbase;
    o.y = (hs * HP + we) * PSTRIDE + bbase;
    o.z = (he * HP + ws) * PSTRIDE + bbase;
    o.w = (he * HP + we) * PSTRIDE + bbase;
    g_off[r][bin] = o;
    int area = (he - hs) * (we - ws);
    g_inv[r][bin] = area > 0 ? 1.f / (float)area : 0.f;
  }
  __syncthreads();

  const int  lane   = tid & 31;
  const int  wv     = tid >> 5;
  const int  r      = lane & 15;
  const int  kbaseA = (lane >= 16) ? 16 : 0;   // 16-bit A 16x32 lane layout (contiguous k = 16h + s)
  const int  kbaseB = (lane >= 16) ? 16 : 0;   // 16-bit B 32x16 lane layout
  const int  o0     = (lane & 15);             // cols 0..15  (all cls)
  const int  o1     = min((lane & 15) + 16, NOUT - 1); // cols 16..24; pad lanes
                                                       // dup row 24 (never read)

  // ---- hoisted per-lane weight row base pointers: addr = base + k0 ----
  // (o*49+bin)*512 + c = o*KTOT + k, so rows are linear in flattened k.
  const float* pB0 = Wc + (size_t)o0 * KTOT + kbaseB;
  const float* pB1 = (o1 < NCLS) ? (Wc + (size_t)o1 * KTOT + kbaseB)
                                 : (Wb + (size_t)(o1 - NCLS) * KTOT + kbaseB);

  v8f acc0 = {}; v8f acc1 = {};

  for (int chunk = wv; chunk < NCHUNK; chunk += 8) {   // 98 iters, uniform per wave
    const int k0  = chunk << 5;
    const int bin = k0 >> 9;        // 512 % 32 == 0: chunk never straddles a bin
    const int c0  = k0 & 511;
    const int4  off = g_off[r][bin];          // one ds_load_b128
    const float inv = g_inv[r][bin];
    const int base = c0 + kbaseA;

    // ---- A fragment: masked bin-average features via 4-corner gathers ----
    v16h a;
    #pragma unroll
    for (int run = 0; run < 2; ++run) {
      const int ro = base + run * 8;
      float s00[8], s01[8], s10[8], s11[8];
      ld8(St + off.x + ro, s00);
      ld8(St + off.y + ro, s01);
      ld8(St + off.z + ro, s10);
      ld8(St + off.w + ro, s11);
      #pragma unroll
      for (int j = 0; j < 8; ++j)
        a[run * 8 + j] = (_Float16)((s11[j] - s01[j] - s10[j] + s00[j]) * inv);
    }

    // ---- B fragments: simple linear loads off hoisted row bases ----
    v16h bf0, bf1;
    {
      float t16[16]; ld16(pB0 + k0, t16);
      #pragma unroll
      for (int i = 0; i < 16; ++i) bf0[i] = (_Float16)t16[i];
    }
    {
      float t16[16]; ld16(pB1 + k0, t16);
      #pragma unroll
      for (int i = 0; i < 16; ++i) bf1[i] = (_Float16)t16[i];
    }

    acc0 = __builtin_amdgcn_wmma_f32_16x16x32_f16(false, a, false, bf0,
                                                  (short)0, acc0, false, false);
    acc1 = __builtin_amdgcn_wmma_f32_16x16x32_f16(false, a, false, bf1,
                                                  (short)0, acc1, false, false);
  }

  // ---- cross-wave reduction of the K-split partial accumulators ----
  #pragma unroll
  for (int v = 0; v < 8; ++v) {
    red[(((wv * 2 + 0) * 8 + v) << 5) + lane] = acc0[v];
    red[(((wv * 2 + 1) * 8 + v) << 5) + lane] = acc1[v];
  }
  __syncthreads();
  for (int e = tid; e < 512; e += 256) {
    int half = e >> 8, rem = e & 255;
    int v = rem >> 5, l = rem & 31;
    float s = 0.f;
    #pragma unroll
    for (int w = 0; w < 8; ++w)
      s += red[(((w * 2 + half) * 8 + v) << 5) + l];
    int M = v + ((l >> 4) << 3);               // C/D f32 16x16 layout
    int N = (l & 15) + (half << 4);
    score[M][N] = s;                           // cols 25..31 = garbage, unread
  }
  __syncthreads();

  // ---- analytic bias term (Σ mask·b[oc]) and bin-mean ----
  for (int e = tid; e < 16 * NOUT; e += 256) {
    int rr = e / NOUT, o = e % NOUT;
    float bs = 0.f;
    for (int bin = 0; bin < 49; ++bin)
      if (g_inv[rr][bin] > 0.f)
        bs += (o < NCLS) ? bc[o * 49 + bin] : bb[(o - NCLS) * 49 + bin];
    score[rr][o] = (score[rr][o] + bs) * (1.f / 49.f);
  }
  __syncthreads();

  // ---- softmax (cls) + direct write (bbox) ----
  if (tid < 16) {
    const int nr = roi0 + tid;
    float mx = score[tid][0];
    for (int o = 1; o < NCLS; ++o) mx = fmaxf(mx, score[tid][o]);
    float ex[NCLS]; float sum = 0.f;
    for (int o = 0; o < NCLS; ++o) { ex[o] = __expf(score[tid][o] - mx); sum += ex[o]; }
    const float is = 1.f / sum;
    for (int o = 0; o < NCLS; ++o) out[(size_t)nr * NCLS + o] = ex[o] * is;
    float* ob = out + (size_t)NROI * NCLS;
    for (int j = 0; j < NBOX; ++j) ob[(size_t)nr * NBOX + j] = score[tid][NCLS + j];
  }
}

extern "C" void kernel_launch(void* const* d_in, const int* in_sizes, int n_in,
                              void* d_out, int out_size, void* d_ws, size_t ws_size,
                              hipStream_t stream) {
  (void)in_sizes; (void)n_in; (void)out_size; (void)ws_size;
  const float* feat = (const float*)d_in[0];
  const float* rois = (const float*)d_in[1];
  const float* Wc   = (const float*)d_in[2];
  const float* bc   = (const float*)d_in[3];
  const float* Wb   = (const float*)d_in[4];
  const float* bb   = (const float*)d_in[5];
  float* out = (float*)d_out;
  float* St  = (float*)d_ws;   // 97*97*2048*4 = 77.1 MB (L2-resident)

  rfcn_integral_kernel<<<PLANES, 128, 0, stream>>>(feat, St);
  rfcn_psroi_wmma_kernel<<<NROI / 16, 256, 0, stream>>>(St, rois, Wc, bc, Wb, bb, out);
}
